// WindowedCrossAttn2D_22359599743007
// MI455X (gfx1250) — matrix operand order repaired and checked by reading, not compiled, not run on hardware
//
#include <hip/hip_runtime.h>

#define C_DIM 192
#define NHEAD 6
#define HD 32
#define WIN 8
#define S_TOK 64
#define WPIX 256
#define HWPIX 65536
#define NWIN 4096  // 4 batches * 32 * 32 windows
#define NW_ELEM (C_DIM * C_DIM)

typedef __attribute__((ext_vector_type(16))) __bf16 v16bf;
typedef __attribute__((ext_vector_type(2)))  __bf16 v2bf;
typedef __attribute__((ext_vector_type(8)))  float  v8f;
typedef __attribute__((ext_vector_type(4)))  int    v4i;
typedef __attribute__((ext_vector_type(4)))  unsigned int u32x4;
typedef __attribute__((ext_vector_type(8)))  unsigned int u32x8;

#if defined(__gfx1250__) && __has_builtin(__builtin_amdgcn_global_load_async_to_lds_b128)
#define HAS_ASYNC_LDS 1
#else
#define HAS_ASYNC_LDS 0
#endif

// 16-byte global -> LDS copy (async DMA on CDNA5 when available)
__device__ __forceinline__ void copy16_g2l(__bf16* dst, const __bf16* src) {
#if HAS_ASYNC_LDS
  __builtin_amdgcn_global_load_async_to_lds_b128(
      (__attribute__((address_space(1))) v4i*)(src),
      (__attribute__((address_space(3))) v4i*)(dst), 0, 0);
#else
  *(u32x4*)dst = *(const u32x4*)src;
#endif
}
__device__ __forceinline__ void wait_async_all() {
#if HAS_ASYNC_LDS
#if __has_builtin(__builtin_amdgcn_s_wait_asynccnt)
  __builtin_amdgcn_s_wait_asynccnt(0);
#endif
#endif
}

// A fragment (16x32 bf16, rows = M) from row-major [row][k], ld in elems.
// Lane m=lane&15 -> row m0+m; elems 0..7 = K 8*half..+7, elems 8..15 = +16.
__device__ __forceinline__ v16bf load_frag_A(const __bf16* s, int m0, int ld, int k0) {
  const int lane = threadIdx.x & 31;
  const __bf16* p = s + (m0 + (lane & 15)) * ld + k0 + 8 * (lane >> 4);
  u32x8 w;
  w.lo = *(const u32x4*)p;
  w.hi = *(const u32x4*)(p + 16);
  return __builtin_bit_cast(v16bf, w);
}

// B fragment (32x16 bf16) with B[k][n] = Wrow[n][k]; Wrow row-major [n][k].
// Lane n=lane&15 -> column n0+n; elems 0..7 = K 8*half..+7, elems 8..15 = +16.
__device__ __forceinline__ v16bf load_frag_Brow(const __bf16* s, int n0, int ld, int k0) {
  const int lane = threadIdx.x & 31;
  const __bf16* p = s + (n0 + (lane & 15)) * ld + k0 + 8 * (lane >> 4);
  u32x8 w;
  w.lo = *(const u32x4*)p;
  w.hi = *(const u32x4*)(p + 16);
  return __builtin_bit_cast(v16bf, w);
}

#define WMMA_BF16(a, b, c) \
  __builtin_amdgcn_wmma_f32_16x16x32_bf16(false, (a), false, (b), (short)0, (c), false, false)

// ---------------------------------------------------------------------------
// Prep: combined weights  Wq_c = wq_i@wq, Wk_c = wk_i@wk, Wv_c = wv_i@wv,
//       Wo_c = proj_w@out_w  (all 192x192, bf16, row-major [out][in])
// combined biases: bq_c = wq_i@bq + bq_i, ..., bo_c = proj_w@out_b + proj_b
// ---------------------------------------------------------------------------
__global__ __launch_bounds__(256)
void prep_weights(const float* __restrict__ wq, const float* __restrict__ bq,
                  const float* __restrict__ wk, const float* __restrict__ bk,
                  const float* __restrict__ wv, const float* __restrict__ bv,
                  const float* __restrict__ in_w, const float* __restrict__ in_b,
                  const float* __restrict__ out_w, const float* __restrict__ out_b,
                  const float* __restrict__ proj_w, const float* __restrict__ proj_b,
                  __bf16* __restrict__ Wc, float* __restrict__ biases) {
  const int idx = blockIdx.x * 256 + threadIdx.x;
  if (idx < 4 * NW_ELEM) {
    const int mat = idx / NW_ELEM;
    const int e = idx - mat * NW_ELEM;
    const int o = e / C_DIM, i = e % C_DIM;
    const float* A; const float* Bm;
    if (mat == 0)      { A = in_w;               Bm = wq; }
    else if (mat == 1) { A = in_w + NW_ELEM;     Bm = wk; }
    else if (mat == 2) { A = in_w + 2 * NW_ELEM; Bm = wv; }
    else               { A = proj_w;             Bm = out_w; }
    float sum = 0.f;
    for (int k = 0; k < C_DIM; ++k) sum += A[o * C_DIM + k] * Bm[k * C_DIM + i];
    Wc[idx] = (__bf16)sum;
  } else if (idx < 4 * NW_ELEM + 4 * C_DIM) {
    const int j = idx - 4 * NW_ELEM;
    const int mat = j / C_DIM, o = j % C_DIM;
    float sum;
    if (mat == 0)      { sum = in_b[o];             for (int k = 0; k < C_DIM; ++k) sum += in_w[o * C_DIM + k] * bq[k]; }
    else if (mat == 1) { sum = in_b[C_DIM + o];     for (int k = 0; k < C_DIM; ++k) sum += in_w[(C_DIM + o) * C_DIM + k] * bk[k]; }
    else if (mat == 2) { sum = in_b[2 * C_DIM + o]; for (int k = 0; k < C_DIM; ++k) sum += in_w[(2 * C_DIM + o) * C_DIM + k] * bv[k]; }
    else               { sum = proj_b[o];           for (int k = 0; k < C_DIM; ++k) sum += proj_w[o * C_DIM + k] * out_b[k]; }
    biases[j] = sum;
  }
}

// ---------------------------------------------------------------------------
// Fused windowed attention: one 8x8 window per block, 8 wave32 per block.
// ---------------------------------------------------------------------------
__global__ __launch_bounds__(256)
void win_attn_kernel(const float* __restrict__ q2d, const float* __restrict__ kv2d,
                     const __bf16* __restrict__ Wc,
                     const float* __restrict__ biases, float* __restrict__ out) {
  // LDS: 147456 B total (two blocks fit in a 320 KB WGP)
  __shared__ __align__(16) __bf16 sX[2 * S_TOK * C_DIM];  // Xq|Xkv -> later P[6][64][64]
  __shared__ __align__(16) __bf16 sQ[S_TOK * C_DIM];      // Q [t][c]
  __shared__ __align__(16) __bf16 sKO[S_TOK * C_DIM];     // K [t][c] -> later O [t][c]
  __shared__ __align__(16) __bf16 sVt[C_DIM * S_TOK];     // V transposed [c][t]
  __shared__ __align__(16) __bf16 sWp[2][C_DIM * 32];     // double-buffered weight K-panels

  const int tid = threadIdx.x;
  const int wave = tid >> 5;
  const int lane = tid & 31;
  const int ln = lane & 15;
  const int lh = lane >> 4;

  // per-wave GEMM mapping: one M-row (m0w), six N-tiles starting at ngrp
  const int m0w = (wave & 3) * 16;
  const int ngrp = (wave >> 2) * 96;

  const int wi = blockIdx.x;
  const int b = wi >> 10;
  const int l = wi & 1023;
  const int hb = l >> 5, wb = l & 31;
  const size_t base = (size_t)b * (C_DIM * (size_t)HWPIX) +
                      (size_t)(hb * WIN) * WPIX + (size_t)(wb * WIN);

  // ---- Phase 0: stage both input windows as bf16 [token][channel] ----------
  for (int j = tid; j < C_DIM * 16; j += 256) {  // 12 iters, float4 along tx
    const int c = j >> 4;
    const int t = (j & 15) << 2;
    const int ty = t >> 3, tx = t & 7;
    const size_t g = base + (size_t)c * HWPIX + (size_t)ty * WPIX + tx;
    const float4 a = *(const float4*)(q2d + g);
    const float4 k4 = *(const float4*)(kv2d + g);
    const int o = t * C_DIM + c;
    sX[o]             = (__bf16)a.x;  sX[o + C_DIM]     = (__bf16)a.y;
    sX[o + 2 * C_DIM] = (__bf16)a.z;  sX[o + 3 * C_DIM] = (__bf16)a.w;
    const int o2 = o + S_TOK * C_DIM;
    sX[o2]             = (__bf16)k4.x;  sX[o2 + C_DIM]     = (__bf16)k4.y;
    sX[o2 + 2 * C_DIM] = (__bf16)k4.z;  sX[o2 + 3 * C_DIM] = (__bf16)k4.w;
  }

  // ---- Phase 1: Q/K/V = X @ Wt + b ----------------------------------------
  for (int mat = 0; mat < 3; ++mat) {
    const __bf16* wsrc = Wc + (size_t)mat * NW_ELEM;
    const __bf16* xsrc = (mat == 0) ? sX : (sX + S_TOK * C_DIM);
    if (mat < 2) __builtin_prefetch(Wc + (size_t)(mat + 1) * NW_ELEM, 0, 2);

    v8f acc[6];
    {
      v8f z = {};
      #pragma unroll
      for (int i = 0; i < 6; ++i) acc[i] = z;
    }
    // prologue: panel 0
    for (int j = tid; j < 768; j += 256) {
      const int o = j >> 2, q = j & 3;
      copy16_g2l(&sWp[0][o * 32 + q * 8], wsrc + o * C_DIM + q * 8);
    }
    wait_async_all();
    __syncthreads();

    for (int ks = 0; ks < 6; ++ks) {
      if (ks < 5) {  // issue next panel copy, overlap with WMMA below
        const __bf16* psrc = wsrc + (ks + 1) * 32;
        __bf16* pdst = sWp[(ks + 1) & 1];
        for (int j = tid; j < 768; j += 256) {
          const int o = j >> 2, q = j & 3;
          copy16_g2l(pdst + o * 32 + q * 8, psrc + o * C_DIM + q * 8);
        }
      }
      const __bf16* pan = sWp[ks & 1];
      v16bf a = load_frag_A(xsrc, m0w, C_DIM, ks * 32);  // shared across 6 tiles
      v16bf bbs[6];
      #pragma unroll
      for (int i = 0; i < 6; ++i) bbs[i] = load_frag_Brow(pan, ngrp + i * 16, 32, 0);
      #pragma unroll
      for (int i = 0; i < 6; ++i) acc[i] = WMMA_BF16(a, bbs[i], acc[i]);
      wait_async_all();
      __syncthreads();
    }

    const float* biasv = biases + mat * C_DIM;
    #pragma unroll
    for (int i = 0; i < 6; ++i) {
      const int n0 = ngrp + i * 16;
      const float bsc = biasv[n0 + ln];
      if (mat == 2) {  // V stored transposed [c][t]: 8 contiguous bf16 per lane
        union { __bf16 us[8]; u32x4 v; } pk;
        #pragma unroll
        for (int r = 0; r < 8; ++r) pk.us[r] = (__bf16)(acc[i][r] + bsc);
        *(u32x4*)(sVt + (n0 + ln) * S_TOK + m0w + 8 * lh) = pk.v;
      } else {
        __bf16* dst = (mat == 0) ? sQ : sKO;
        #pragma unroll
        for (int r = 0; r < 8; ++r)
          dst[(m0w + 8 * lh + r) * C_DIM + n0 + ln] = (__bf16)(acc[i][r] + bsc);
      }
    }
  }
  __syncthreads();

  // ---- Phase 2: scores S_h = Q_h @ K_h^T  (6 heads x 4m x 4n, K=32) --------
  #pragma unroll
  for (int ic = 0; ic < 3; ++ic) {
    const int combo = wave * 3 + ic;      // (head, m-tile): 24 combos
    const int head = combo >> 2;
    const int m0 = (combo & 3) * 16;
    v16bf a = load_frag_A(sQ, m0, C_DIM, head * HD);  // shared across 4 n-tiles
    v16bf bbs[4];
    #pragma unroll
    for (int j = 0; j < 4; ++j)
      bbs[j] = load_frag_Brow(sKO, j * 16, C_DIM, head * HD);  // B[k][n]=K[n][k]
    v8f accs[4];
    #pragma unroll
    for (int j = 0; j < 4; ++j) {
      v8f acc = {};
      accs[j] = WMMA_BF16(a, bbs[j], acc);
    }
    __bf16* P = sX + head * (S_TOK * S_TOK);
    #pragma unroll
    for (int j = 0; j < 4; ++j)
      #pragma unroll
      for (int r = 0; r < 8; ++r)
        P[(m0 + 8 * lh + r) * S_TOK + j * 16 + ln] = (__bf16)accs[j][r];
  }
  __syncthreads();

  // ---- Phase 3: softmax over 384 rows (scale 1/sqrt(32)), 2-wide -----------
  for (int row = tid; row < NHEAD * S_TOK; row += 256) {
    v2bf* p2 = (v2bf*)(sX + row * S_TOK);
    float v[64];
    float mx = -3.4e38f;
    for (int j = 0; j < 32; ++j) {
      const v2bf t = p2[j];
      const float x0 = (float)t.x * 0.17677669529663689f;
      const float x1 = (float)t.y * 0.17677669529663689f;
      v[2 * j] = x0; v[2 * j + 1] = x1;
      mx = fmaxf(mx, fmaxf(x0, x1));
    }
    float sum = 0.f;
    for (int j = 0; j < 64; ++j) { const float e = __expf(v[j] - mx); v[j] = e; sum += e; }
    const float inv = 1.0f / sum;
    for (int j = 0; j < 32; ++j) {
      v2bf t;
      t.x = (__bf16)(v[2 * j] * inv);
      t.y = (__bf16)(v[2 * j + 1] * inv);
      p2[j] = t;
    }
  }
  __syncthreads();

  // ---- Phase 4: O_h = P_h @ V_h  (6 heads x 4m x 2n, K=64) -----------------
  #pragma unroll
  for (int ic = 0; ic < 3; ++ic) {
    const int combo = wave * 3 + ic;
    const int head = combo >> 2;
    const int m0 = (combo & 3) * 16;
    const __bf16* P = sX + head * (S_TOK * S_TOK);
    v16bf as[2], bbs[4];
    #pragma unroll
    for (int ks = 0; ks < 2; ++ks) {
      as[ks] = load_frag_A(P, m0, S_TOK, ks * 32);
      #pragma unroll
      for (int j = 0; j < 2; ++j)
        bbs[ks * 2 + j] = load_frag_Brow(sVt, head * HD + j * 16, S_TOK, ks * 32);
    }
    v8f acc[2];
    { v8f z = {}; acc[0] = z; acc[1] = z; }
    #pragma unroll
    for (int ks = 0; ks < 2; ++ks)
      #pragma unroll
      for (int j = 0; j < 2; ++j)
        acc[j] = WMMA_BF16(as[ks], bbs[ks * 2 + j], acc[j]);
    #pragma unroll
    for (int j = 0; j < 2; ++j)
      #pragma unroll
      for (int r = 0; r < 8; ++r)
        sKO[(m0 + 8 * lh + r) * C_DIM + head * HD + j * 16 + ln] = (__bf16)acc[j][r];
  }

  // ---- Phase 5: Out = O @ Wo_c^T + bo + residual ---------------------------
  v8f facc[6];
  {
    v8f z = {};
    #pragma unroll
    for (int i = 0; i < 6; ++i) facc[i] = z;
  }
  const __bf16* wosrc = Wc + (size_t)3 * NW_ELEM;
  // prologue panel 0 (barrier below also publishes phase-4 O stores)
  for (int j = tid; j < 768; j += 256) {
    const int o = j >> 2, q = j & 3;
    copy16_g2l(&sWp[0][o * 32 + q * 8], wosrc + o * C_DIM + q * 8);
  }
  wait_async_all();
  __syncthreads();

  for (int ks = 0; ks < 6; ++ks) {
    if (ks < 5) {
      const __bf16* psrc = wosrc + (ks + 1) * 32;
      __bf16* pdst = sWp[(ks + 1) & 1];
      for (int j = tid; j < 768; j += 256) {
        const int o = j >> 2, q = j & 3;
        copy16_g2l(pdst + o * 32 + q * 8, psrc + o * C_DIM + q * 8);
      }
    }
    const __bf16* pan = sWp[ks & 1];
    v16bf a = load_frag_A(sKO, m0w, C_DIM, ks * 32);
    v16bf bbs[6];
    #pragma unroll
    for (int i = 0; i < 6; ++i) bbs[i] = load_frag_Brow(pan, ngrp + i * 16, 32, 0);
    #pragma unroll
    for (int i = 0; i < 6; ++i) facc[i] = WMMA_BF16(a, bbs[i], facc[i]);
    wait_async_all();
    __syncthreads();
  }

  const float* biasO = biases + 3 * C_DIM;
  #pragma unroll
  for (int i = 0; i < 6; ++i) {
    const int ch = ngrp + i * 16 + ln;
    const int mb = m0w + 8 * lh;          // token base, multiple of 8 -> one ty row
    const int ty = mb >> 3;
    const float bsc = biasO[ch];
    const size_t g = base + (size_t)ch * HWPIX + (size_t)ty * WPIX;  // +r for tx
    const float4 r0 = *(const float4*)(q2d + g);
    const float4 r1 = *(const float4*)(q2d + g + 4);
    float4 o0, o1;
    o0.x = facc[i][0] + bsc + r0.x;  o0.y = facc[i][1] + bsc + r0.y;
    o0.z = facc[i][2] + bsc + r0.z;  o0.w = facc[i][3] + bsc + r0.w;
    o1.x = facc[i][4] + bsc + r1.x;  o1.y = facc[i][5] + bsc + r1.y;
    o1.z = facc[i][6] + bsc + r1.z;  o1.w = facc[i][7] + bsc + r1.w;
    *(float4*)(out + g) = o0;
    *(float4*)(out + g + 4) = o1;
  }
}

extern "C" void kernel_launch(void* const* d_in, const int* in_sizes, int n_in,
                              void* d_out, int out_size, void* d_ws, size_t ws_size,
                              hipStream_t stream) {
  const float* q2d    = (const float*)d_in[0];
  const float* kv2d   = (const float*)d_in[1];
  const float* wq     = (const float*)d_in[2];
  const float* bq     = (const float*)d_in[3];
  const float* wk     = (const float*)d_in[4];
  const float* bk     = (const float*)d_in[5];
  const float* wv     = (const float*)d_in[6];
  const float* bv     = (const float*)d_in[7];
  const float* in_w   = (const float*)d_in[8];
  const float* in_b   = (const float*)d_in[9];
  const float* out_w  = (const float*)d_in[10];
  const float* out_b  = (const float*)d_in[11];
  const float* proj_w = (const float*)d_in[12];
  const float* proj_b = (const float*)d_in[13];
  float* out = (float*)d_out;

  __bf16* Wc = (__bf16*)d_ws;                                          // 4 * 192*192 bf16
  float* biases = (float*)((char*)d_ws + (size_t)4 * NW_ELEM * 2);     // 4 * 192 f32

  const int nprep = 4 * NW_ELEM + 4 * C_DIM;
  prep_weights<<<(nprep + 255) / 256, 256, 0, stream>>>(
      wq, bq, wk, bk, wv, bv, in_w, in_b, out_w, out_b, proj_w, proj_b, Wc, biases);
  win_attn_kernel<<<NWIN, 256, 0, stream>>>(q2d, kv2d, Wc, biases, out);
}
